// Critic_13125420056616
// MI455X (gfx1250) — matrix operand order repaired and checked by reading, not compiled, not run on hardware
//
#include <hip/hip_runtime.h>
#include <hip/hip_bf16.h>

typedef _Float16 f16;
typedef _Float16 f16x8  __attribute__((ext_vector_type(8)));
typedef _Float16 f16x16 __attribute__((ext_vector_type(16)));
typedef float    f32x8  __attribute__((ext_vector_type(8)));
typedef unsigned int u32x4 __attribute__((ext_vector_type(4)));
typedef int          i32x8 __attribute__((ext_vector_type(8)));
typedef int          i32x4 __attribute__((ext_vector_type(4)));

static __device__ __forceinline__ f16x16 cat16(f16x8 a, f16x8 b) {
  return __builtin_shufflevector(a, b, 0,1,2,3,4,5,6,7,8,9,10,11,12,13,14,15);
}

static __device__ __forceinline__ f32x8 wmma32(f16x16 a, f16x16 b, f32x8 c) {
  // D(f32 16x16) = A(f16 16x32) * B(f16 32x16) + C
  return __builtin_amdgcn_wmma_f32_16x16x32_f16(false, a, false, b, (short)0, c,
                                                false, false);
}

// ---------------------------------------------------------------- utilities

__global__ void zero_f32(float* __restrict__ p, size_t n) {
  size_t i = (size_t)blockIdx.x * blockDim.x + threadIdx.x;
  if (i < n) p[i] = 0.0f;
}

// Bt[n][k] = W[k][n]  (f32 [K,Nc] row-major -> f16 [Nc,K] row-major)
__global__ void transpose_f16(const float* __restrict__ W, f16* __restrict__ Bt,
                              int K, int Nc) {
  int i = blockIdx.x * blockDim.x + threadIdx.x;
  if (i >= K * Nc) return;
  int n = i / K, k = i - n * K;
  Bt[i] = (f16)W[(size_t)k * Nc + n];
}

// Split-concat weight: W is [2K, H]. Output Bt is [2H, K] f16 with
//   Bt[n][k] = W[k][n]       for n <  H   (dst half)
//   Bt[n][k] = W[K+k][n-H]   for n >= H   (src half)
__global__ void transpose_cat_f16(const float* __restrict__ W, f16* __restrict__ Bt,
                                  int K, int H) {
  int i = blockIdx.x * blockDim.x + threadIdx.x;
  if (i >= 2 * H * K) return;
  int n = i / K, k = i - n * K;
  float v = (n < H) ? W[(size_t)k * H + n] : W[(size_t)(K + k) * H + (n - H)];
  Bt[i] = (f16)v;
}

// out[i] = b[i] for i < H else 0   (length L = 2H)
__global__ void bias_cat(const float* __restrict__ b, float* __restrict__ out,
                         int H, int L) {
  int i = threadIdx.x;
  if (i < L) out[i] = (i < H) ? b[i] : 0.0f;
}

// nf[n][c] = c < 96 ? x[n][c] : action[n/64][c-96]   (f16, [N,128])
__global__ void build_nf(const float* __restrict__ x, const float* __restrict__ act,
                         f16* __restrict__ nf, int N) {
  int i = blockIdx.x * blockDim.x + threadIdx.x;
  if (i >= N * 128) return;
  int n = i >> 7, c = i & 127;
  float v = (c < 96) ? x[(size_t)n * 96 + c]
                     : act[(size_t)(n >> 6) * 32 + (c - 96)];
  nf[i] = (f16)v;
}

// ---------------------------------------------------------------- WMMA GEMM
// C[M,Nc](f16) = relu?( A[M,K](f16) @ Bt[Nc,K]^T + bias[Nc] )
// One wave = 16 rows x 64 cols (4 N-tiles share each A fragment load).
template<int RELU>
__global__ void __launch_bounds__(128)
wmma_gemm(const f16* __restrict__ A, const f16* __restrict__ Bt,
          const float* __restrict__ bias, f16* __restrict__ C,
          int M, int K, int Nc) {
  constexpr int NTW = 4;
  const int lane = threadIdx.x & 31;
  const int wave = blockIdx.x * (blockDim.x >> 5) + (threadIdx.x >> 5);
  const int ngrp = Nc >> 6;                       // groups of 4 N-tiles
  const int mt = wave / ngrp, ng = wave - mt * ngrp;
  if (mt * 16 >= M) return;
  const int l15 = lane & 15, hi = lane >> 4;
  const size_t arow = (size_t)(mt * 16 + l15) * K;
  const int n0 = ng * 64 + l15;
  f32x8 acc[NTW];
  const f32x8 zero = {0.f, 0.f, 0.f, 0.f, 0.f, 0.f, 0.f, 0.f};
#pragma unroll
  for (int t = 0; t < NTW; ++t) acc[t] = zero;
  for (int kk = 0; kk < K; kk += 32) {
    // A 16x32 f16 layout: lanes<16 hold K {0..7,16..23}; lanes>=16 {8..15,24..31}
    const f16* ap = A + arow + kk + hi * 8;
    f16x16 af = cat16(*(const f16x8*)ap, *(const f16x8*)(ap + 16));
#pragma unroll
    for (int t = 0; t < NTW; ++t) {
      // B 32x16 f16 layout: lanes<16 hold K {0..7,16..23}; lanes>=16 {8..15,24..31} (col = l15)
      const f16* bp = Bt + (size_t)(n0 + t * 16) * K + kk + hi * 8;
      acc[t] = wmma32(af, cat16(*(const f16x8*)bp, *(const f16x8*)(bp + 16)), acc[t]);
    }
  }
#pragma unroll
  for (int t = 0; t < NTW; ++t) {
    const int ncol = n0 + t * 16;
    const float bv = bias[ncol];
#pragma unroll
    for (int v = 0; v < 8; ++v) {
      float cv = acc[t][v] + bv;                  // C/D: lane row m = v + 8*hi
      if (RELU) cv = cv > 0.f ? cv : 0.f;
      C[(size_t)(mt * 16 + v + hi * 8) * Nc + ncol] = (f16)cv;
    }
  }
}

// ------------------------------------------------------------ edge message MLP
// Per 16-edge tile (one wave32):
//   TDM gather-mode DMA pulls AB[dst[m]][0:KD] and AB[src[m]][KD:2KD] into LDS,
//   pre[m][k] = relu(dst_row[k] + src_row[k])  (b1 pre-folded into AB)
//   msg = pre @ W2 (WMMA f32 accum), scatter: accum[dst[m]][n] += msg[m][n]
template<int KD, int NC>
__global__ void __launch_bounds__(32)
edge_wmma(const int* __restrict__ srcs, const int* __restrict__ dsts,
          const f16* __restrict__ ABn,   // [N, 2*KD] f16: A(+bias) | B halves
          const f16* __restrict__ Wt,    // [NC, KD]  f16 (transposed W2)
          float* __restrict__ accum,     // [N, NC]   f32
          int E) {
  __shared__ __align__(128) f16 gbuf[2][16 * KD];   // [0]=dst rows, [1]=src rows
  const int lane = threadIdx.x;
  const int l15  = lane & 15;
  const int hi   = lane >> 4;
  const int e0   = blockIdx.x * 16;
  if (e0 + 31 < E) {                      // speculative prefetch of next tile ids
    __builtin_prefetch(dsts + e0 + 16, 0, 1);
    __builtin_prefetch(srcs + e0 + 16, 0, 1);
  }
  // lanes 0..15 carry the 16 dst ids, lanes 16..31 the 16 src ids
  const int myid = (lane < 16) ? dsts[e0 + l15] : srcs[e0 + l15];

  // pack 16-bit row indices into wave-uniform words for the TDM descriptor
  int dw[8], sw[8];
#pragma unroll
  for (int j = 0; j < 8; ++j) {
    int a0 = __builtin_amdgcn_readlane(myid, 2 * j);
    int a1 = __builtin_amdgcn_readlane(myid, 2 * j + 1);
    dw[j] = (a0 & 0xffff) | (a1 << 16);
    int b0 = __builtin_amdgcn_readlane(myid, 16 + 2 * j);
    int b1 = __builtin_amdgcn_readlane(myid, 16 + 2 * j + 1);
    sw[j] = (b0 & 0xffff) | (b1 << 16);
  }

  // ---- Tensor DMA descriptors (gather mode, 16-bit indices) ----
  const unsigned lds0 = (unsigned)(size_t)&gbuf[0][0];
  const unsigned lds1 = (unsigned)(size_t)&gbuf[1][0];
  const unsigned long long ga = (unsigned long long)(size_t)ABn;        // A half
  const unsigned long long gb = (unsigned long long)(size_t)(ABn + KD); // B half
  // group0: count=1, gather_mode=1 (bit31), 16-bit idx (bit30=0), lds, gaddr, type=2
  u32x4 g0a = {1u | 0x80000000u, lds0, (unsigned)(ga & 0xffffffffu),
               (unsigned)((ga >> 32) & 0x01ffffffu) | (2u << 30)};
  u32x4 g0b = {1u | 0x80000000u, lds1, (unsigned)(gb & 0xffffffffu),
               (unsigned)((gb >> 32) & 0x01ffffffu) | (2u << 30)};
  // group1: data_size=2B, tensor_dim0=2*KD, tensor_dim1=65536,
  //         tile_dim0=KD, tile_dim1=16 indices, dim0 stride=2*KD
  i32x8 g1 = {(int)(1u << 16),
              (int)((unsigned)(2 * KD) << 16),
              0,
              (int)(1u | ((unsigned)KD << 16)),
              16,
              (int)(2 * KD),
              0, 0};
  i32x4 g2a = {dw[0], dw[1], dw[2], dw[3]};
  i32x4 g3a = {dw[4], dw[5], dw[6], dw[7]};
  i32x4 g2b = {sw[0], sw[1], sw[2], sw[3]};
  i32x4 g3b = {sw[4], sw[5], sw[6], sw[7]};
  i32x8 g23a = {dw[0], dw[1], dw[2], dw[3], dw[4], dw[5], dw[6], dw[7]};
  i32x8 g23b = {sw[0], sw[1], sw[2], sw[3], sw[4], sw[5], sw[6], sw[7]};

  __builtin_amdgcn_tensor_load_to_lds(g0a, g1, g2a, g3a, g23a, 0);
  __builtin_amdgcn_tensor_load_to_lds(g0b, g1, g2b, g3b, g23b, 0);
  __builtin_amdgcn_s_wait_tensorcnt(0);
  __syncthreads();

  // build A fragments once (register-cached across all N-tiles): add + ReLU
  f16x16 afrag[KD / 32];
#pragma unroll
  for (int kk = 0; kk < KD; kk += 32) {
    const f16* pa = &gbuf[0][l15 * KD + kk + hi * 8];
    const f16* pb = &gbuf[1][l15 * KD + kk + hi * 8];
    f16x16 va = cat16(*(const f16x8*)pa, *(const f16x8*)(pa + 16));
    f16x16 vb = cat16(*(const f16x8*)pb, *(const f16x8*)(pb + 16));
    f16x16 s = va + vb;
#pragma unroll
    for (int i = 0; i < 16; ++i) s[i] = s[i] > (f16)0 ? s[i] : (f16)0;
    afrag[kk / 32] = s;
  }

#pragma unroll
  for (int nt = 0; nt < NC / 16; ++nt) {
    f32x8 acc = {0.f, 0.f, 0.f, 0.f, 0.f, 0.f, 0.f, 0.f};
#pragma unroll
    for (int kk = 0; kk < KD; kk += 32) {
      const f16* bp = Wt + (size_t)(nt * 16 + l15) * KD + kk + hi * 8;
      acc = wmma32(afrag[kk / 32], cat16(*(const f16x8*)bp, *(const f16x8*)(bp + 16)), acc);
    }
    const int n = nt * 16 + l15;
#pragma unroll
    for (int v = 0; v < 8; ++v) {
      const int m = v + hi * 8;                  // edge row of this partial
      const int dm = __shfl(myid, m, 32);        // dst id held by lane m
      atomicAdd(accum + (size_t)dm * NC + n, acc[v]);
    }
  }
}

__global__ void count_edges(const int* __restrict__ dsts, float* __restrict__ cnt,
                            int E) {
  int e = blockIdx.x * blockDim.x + threadIdx.x;
  if (e < E) atomicAdd(cnt + dsts[e], 1.0f);
}

// h[n][c] = relu( cnt>0 ? acc[n][c]/cnt + b2[c] : 0 )  -> f16
__global__ void finalize_relu_f16(const float* __restrict__ acc,
                                  const float* __restrict__ cnt,
                                  const float* __restrict__ b2,
                                  f16* __restrict__ out, int N, int C) {
  int i = blockIdx.x * blockDim.x + threadIdx.x;
  if (i >= N * C) return;
  int n = i / C, c = i - n * C;
  float k = cnt[n];
  float v = (k > 0.f) ? (acc[i] / k + b2[c]) : 0.f;
  out[i] = (f16)(v > 0.f ? v : 0.f);
}

// h[n][c] = cnt>0 ? acc[n][c]/cnt + b2[c] : 0  -> f32 (no relu, layer-2 output)
__global__ void finalize_f32(const float* __restrict__ acc,
                             const float* __restrict__ cnt,
                             const float* __restrict__ b2,
                             float* __restrict__ out, int N, int C) {
  int i = blockIdx.x * blockDim.x + threadIdx.x;
  if (i >= N * C) return;
  int n = i / C, c = i - n * C;
  float k = cnt[n];
  out[i] = (k > 0.f) ? (acc[i] / k + b2[c]) : 0.f;
}

// ge[g][c] = mean over 64 contiguous nodes (batch = node/64)
__global__ void graph_mean(const float* __restrict__ h2, float* __restrict__ ge) {
  int g = blockIdx.x, c = threadIdx.x;
  float s = 0.f;
  for (int i = 0; i < 64; ++i) s += h2[((size_t)g * 64 + i) * 64 + c];
  ge[(size_t)g * 64 + c] = s * (1.0f / 64.0f);
}

// z[g] = cat(state[96], action[32], ge[64]) -> f16 [B,192]
__global__ void build_z(const float* __restrict__ st, const float* __restrict__ ac,
                        const float* __restrict__ ge, f16* __restrict__ z) {
  int g = blockIdx.x, t = threadIdx.x;
  float v = (t < 96)  ? st[(size_t)g * 96 + t]
          : (t < 128) ? ac[(size_t)g * 32 + (t - 96)]
                      : ge[(size_t)g * 64 + (t - 128)];
  z[(size_t)g * 192 + t] = (f16)v;
}

// q[g] = z2[g,:] . w3 + b3     (final 256 -> 1 layer)
__global__ void qhead_out(const f16* __restrict__ z2, const float* __restrict__ w3,
                          const float* __restrict__ b3, float* __restrict__ out,
                          int Bn) {
  int g = blockIdx.x * blockDim.x + threadIdx.x;
  if (g >= Bn) return;
  float s = b3[0];
  const f16* r = z2 + (size_t)g * 256;
  for (int k = 0; k < 256; ++k) s += (float)r[k] * w3[k];
  out[g] = s;
}

// ---------------------------------------------------------------- launcher

extern "C" void kernel_launch(void* const* d_in, const int* in_sizes, int n_in,
                              void* d_out, int out_size, void* d_ws, size_t ws_size,
                              hipStream_t stream) {
  (void)in_sizes; (void)n_in; (void)out_size; (void)ws_size;
  constexpr int N = 65536, E = 524288, B = 1024;

  const float* state  = (const float*)d_in[0];
  const float* action = (const float*)d_in[1];
  const float* x      = (const float*)d_in[2];
  const int*   ei     = (const int*)d_in[3];
  const float* g1_w1  = (const float*)d_in[5];
  const float* g1_b1  = (const float*)d_in[6];
  const float* g1_w2  = (const float*)d_in[7];
  const float* g1_b2  = (const float*)d_in[8];
  const float* g2_w1  = (const float*)d_in[9];
  const float* g2_b1  = (const float*)d_in[10];
  const float* g2_w2  = (const float*)d_in[11];
  const float* g2_b2  = (const float*)d_in[12];
  const float* q1_w1  = (const float*)d_in[13]; const float* q1_b1 = (const float*)d_in[14];
  const float* q1_w2  = (const float*)d_in[15]; const float* q1_b2 = (const float*)d_in[16];
  const float* q1_w3  = (const float*)d_in[17]; const float* q1_b3 = (const float*)d_in[18];
  const float* q2_w1  = (const float*)d_in[19]; const float* q2_b1 = (const float*)d_in[20];
  const float* q2_w2  = (const float*)d_in[21]; const float* q2_b2 = (const float*)d_in[22];
  const float* q2_w3  = (const float*)d_in[23]; const float* q2_b3 = (const float*)d_in[24];
  const int* srcs = ei;        // edge_index[0]
  const int* dsts = ei + E;    // edge_index[1]
  float* out = (float*)d_out;

  char* w = (char*)d_ws;
  auto alloc = [&](size_t bytes) -> void* {
    void* p = (void*)w;
    w += (bytes + 255) & ~(size_t)255;
    return p;
  };
  f16*   nf    = (f16*)  alloc((size_t)N * 128 * 2);   // node features f16
  f16*   AB1   = (f16*)  alloc((size_t)N * 256 * 2);   // layer1 per-node A(+b1)|B
  float* acc1  = (float*)alloc((size_t)N * 128 * 4);   // layer1 segment sums
  float* acc2  = (float*)alloc((size_t)N * 64 * 4);    // layer2 segment sums
  float* cnt   = (float*)alloc((size_t)N * 4);         // in-degree counts
  f16*   h1    = (f16*)  alloc((size_t)N * 128 * 2);   // layer1 node output
  f16*   AB2   = (f16*)  alloc((size_t)N * 128 * 2);   // layer2 per-node A(+b1)|B
  float* h2    = (float*)alloc((size_t)N * 64 * 4);    // layer2 node output
  float* ge    = (float*)alloc((size_t)B * 64 * 4);    // graph embeddings
  f16*   z     = (f16*)  alloc((size_t)B * 192 * 2);   // q-head input
  f16*   t1    = (f16*)  alloc((size_t)B * 256 * 2);   // q-head hidden 1
  f16*   t2    = (f16*)  alloc((size_t)B * 256 * 2);   // q-head hidden 2
  f16*   Bt1   = (f16*)  alloc(256 * 128 * 2);         // layer1 W1 split+T
  float* bias1 = (float*)alloc(256 * 4);
  f16*   Bt2   = (f16*)  alloc(128 * 128 * 2);         // layer1 W2^T
  f16*   Bt3   = (f16*)  alloc(128 * 128 * 2);         // layer2 W1 split+T
  float* bias3 = (float*)alloc(128 * 4);
  f16*   Bt4   = (f16*)  alloc(64 * 64 * 2);           // layer2 W2^T
  f16*   Bq11  = (f16*)  alloc(256 * 192 * 2);
  f16*   Bq12  = (f16*)  alloc(256 * 256 * 2);
  f16*   Bq21  = (f16*)  alloc(256 * 192 * 2);
  f16*   Bq22  = (f16*)  alloc(256 * 256 * 2);

  // zero accumulators (acc1|acc2|cnt are contiguous, sizes are 256B multiples)
  {
    size_t nz = (size_t)N * (128 + 64 + 1);
    zero_f32<<<(unsigned)((nz + 255) / 256), 256, 0, stream>>>(acc1, nz);
  }
  // weight prep: f16, [N][K] layouts for WMMA B operand
  transpose_cat_f16<<<(2 * 128 * 128 + 255) / 256, 256, 0, stream>>>(g1_w1, Bt1, 128, 128);
  transpose_cat_f16<<<(2 * 64 * 128 + 255) / 256, 256, 0, stream>>>(g2_w1, Bt3, 128, 64);
  transpose_f16<<<(128 * 128 + 255) / 256, 256, 0, stream>>>(g1_w2, Bt2, 128, 128);
  transpose_f16<<<(64 * 64 + 255) / 256, 256, 0, stream>>>(g2_w2, Bt4, 64, 64);
  transpose_f16<<<(192 * 256 + 255) / 256, 256, 0, stream>>>(q1_w1, Bq11, 192, 256);
  transpose_f16<<<(256 * 256 + 255) / 256, 256, 0, stream>>>(q1_w2, Bq12, 256, 256);
  transpose_f16<<<(192 * 256 + 255) / 256, 256, 0, stream>>>(q2_w1, Bq21, 192, 256);
  transpose_f16<<<(256 * 256 + 255) / 256, 256, 0, stream>>>(q2_w2, Bq22, 256, 256);
  bias_cat<<<1, 256, 0, stream>>>(g1_b1, bias1, 128, 256);
  bias_cat<<<1, 128, 0, stream>>>(g2_b1, bias3, 64, 128);

  // node features [N,128] f16
  build_nf<<<(N * 128 + 255) / 256, 256, 0, stream>>>(x, action, nf, N);

  // ---- GNN layer 1: node precompute, edge MLP (TDM gather), mean+bias+relu
  wmma_gemm<0><<<(N / 16) * (256 / 64) / 4, 128, 0, stream>>>(nf, Bt1, bias1, AB1,
                                                              N, 128, 256);
  edge_wmma<128, 128><<<E / 16, 32, 0, stream>>>(srcs, dsts, AB1, Bt2, acc1, E);
  count_edges<<<E / 256, 256, 0, stream>>>(dsts, cnt, E);
  finalize_relu_f16<<<(N * 128 + 255) / 256, 256, 0, stream>>>(acc1, cnt, g1_b2,
                                                               h1, N, 128);
  // ---- GNN layer 2
  wmma_gemm<0><<<(N / 16) * (128 / 64) / 4, 128, 0, stream>>>(h1, Bt3, bias3, AB2,
                                                              N, 128, 128);
  edge_wmma<64, 64><<<E / 16, 32, 0, stream>>>(srcs, dsts, AB2, Bt4, acc2, E);
  finalize_f32<<<(N * 64 + 255) / 256, 256, 0, stream>>>(acc2, cnt, g2_b2, h2, N, 64);

  // ---- readout + q-heads
  graph_mean<<<B, 64, 0, stream>>>(h2, ge);
  build_z<<<B, 192, 0, stream>>>(state, action, ge, z);

  wmma_gemm<1><<<(B / 16) * (256 / 64) / 4, 128, 0, stream>>>(z, Bq11, q1_b1, t1,
                                                              B, 192, 256);
  wmma_gemm<1><<<(B / 16) * (256 / 64) / 4, 128, 0, stream>>>(t1, Bq12, q1_b2, t2,
                                                              B, 256, 256);
  qhead_out<<<B / 256, 256, 0, stream>>>(t2, q1_w3, q1_b3, out, B);

  wmma_gemm<1><<<(B / 16) * (256 / 64) / 4, 128, 0, stream>>>(z, Bq21, q2_b1, t1,
                                                              B, 192, 256);
  wmma_gemm<1><<<(B / 16) * (256 / 64) / 4, 128, 0, stream>>>(t1, Bq22, q2_b2, t2,
                                                              B, 256, 256);
  qhead_out<<<B / 256, 256, 0, stream>>>(t2, q2_w3, q2_b3, out + B, B);
}
